// LSTMWithAttention_30897994727671
// MI455X (gfx1250) — matrix operand order repaired and checked by reading, not compiled, not run on hardware
//
#include <hip/hip_runtime.h>
#include <math.h>

// ---------------------------------------------------------------------------
// B=16, T=256, I=256, HE=512, V=128, HD=512
// WMMA bf16 16x16x32, f32 accumulate. wave32. Double-buffered LDS stages.
// ---------------------------------------------------------------------------

typedef float  v8f   __attribute__((ext_vector_type(8)));
typedef __bf16 v16bf __attribute__((ext_vector_type(16)));

union Frag { v16bf v; unsigned int u[8]; };

// round-to-nearest-even fp32 -> bf16, pack two into a dword (lo = even K)
__device__ __forceinline__ unsigned int pk_bf16(float a, float b) {
    unsigned int ua = __float_as_uint(a);
    unsigned int ub = __float_as_uint(b);
    ua += 0x7FFFu + ((ua >> 16) & 1u);
    ub += 0x7FFFu + ((ub >> 16) & 1u);
    return (ua >> 16) | (ub & 0xFFFF0000u);
}

__device__ __forceinline__ float sigmoidf_(float x) {
    return 1.0f / (1.0f + __expf(-x));
}

// LDS tile pitch in dwords (16 data dwords = 32 bf16 K-values, +4 pad)
#define TPITCH 20

// 64-row x 32-K tile loader: 256 threads, 8 floats each (two float4 loads).
__device__ __forceinline__ void load_tile64(unsigned int* Lt, const float* A,
                                            int lda, int rbase, int k0, int tid) {
    int row = tid >> 2;      // 0..63
    int seg = tid & 3;       // 0..3
    const float* src = A + (size_t)(rbase + row) * lda + k0 + seg * 8;
    float4 f0 = *(const float4*)(src);
    float4 f1 = *(const float4*)(src + 4);
    unsigned int* dst = Lt + row * TPITCH + seg * 4;
    dst[0] = pk_bf16(f0.x, f0.y);
    dst[1] = pk_bf16(f0.z, f0.w);
    dst[2] = pk_bf16(f1.x, f1.y);
    dst[3] = pk_bf16(f1.z, f1.w);
}

// 32-row x 32-K tile loader: 256 threads, 4 floats each.
__device__ __forceinline__ void load_tile32(unsigned int* Lt, const float* W,
                                            int ldw, int rbase, int k0, int tid) {
    int row = tid >> 3;      // 0..31
    int seg = tid & 7;       // 0..7
    const float* src = W + (size_t)(rbase + row) * ldw + k0 + seg * 4;
    float4 f = *(const float4*)(src);
    unsigned int* dst = Lt + row * TPITCH + seg * 2;
    dst[0] = pk_bf16(f.x, f.y);
    dst[1] = pk_bf16(f.z, f.w);
}

__device__ __forceinline__ void prefetch_row(const float* A, int lda, int rbase,
                                             int k0, int tid) {
    int row = tid >> 2;
    int seg = tid & 3;
    __builtin_prefetch(A + (size_t)(rbase + row) * lda + k0 + seg * 8, 0, 0);
}

// A fragment per ISA 7.12.2 (16-bit A 16x32):
//   lanes 0-15:  VGPR0-3 -> K 0..7,  VGPR4-7 -> K 16..23
//   lanes 16-31: VGPR0-3 -> K 8..15, VGPR4-7 -> K 24..31
__device__ __forceinline__ v16bf build_a(const unsigned int* At, int m_off, int lane) {
    Frag f;
    const unsigned int* p = At + (m_off + (lane & 15)) * TPITCH;
    int h4 = (lane >> 4) * 4;
#pragma unroll
    for (int j = 0; j < 4; ++j) f.u[j] = p[h4 + j];
#pragma unroll
    for (int j = 0; j < 4; ++j) f.u[4 + j] = p[8 + h4 + j];
    return f.v;
}

// B fragment (16-bit B 32x16): lane = N col; lanes 0-15 hold K 0..7,16..23,
// lanes 16-31 hold K 8..15,24..31, two K per VGPR. Tile stored [n][k].
__device__ __forceinline__ v16bf build_b(const unsigned int* Bt, int n_off, int lane) {
    Frag f;
    const unsigned int* p = Bt + (n_off + (lane & 15)) * TPITCH + (lane >> 4) * 4;
#pragma unroll
    for (int j = 0; j < 8; ++j) f.u[j] = p[(j & 3) + (j >> 2) * 8];
    return f.v;
}

#define WMMA_BF16(a, b, c) \
    __builtin_amdgcn_wmma_f32_16x16x32_bf16(false, (a), false, (b), (short)0, (c), false, false)

// ---------------------------------------------------------------------------
// Generic C[M,N] = A[M,K] @ W[N,K]^T.  Block tile 64x64, 8 waves (4x2),
// wave tile 16x32 (2 accumulators, shared A fragment), double-buffered LDS.
// grid = (M/64, N/64)
// ---------------------------------------------------------------------------
__global__ void gemm_wT_wmma(const float* __restrict__ A, int lda,
                             const float* __restrict__ W, int ldw,
                             float* __restrict__ C, int ldc, int K) {
    __shared__ unsigned int At[2][64 * TPITCH];
    __shared__ unsigned int Bt[2][64 * TPITCH];
    int tid  = threadIdx.x;
    int lane = tid & 31;
    int wv   = tid >> 5;
    int mbase = blockIdx.x * 64;
    int nbase = blockIdx.y * 64;
    int m_off = (wv & 3) * 16;
    int n_off = (wv >> 2) * 32;

    const int S = K >> 5;
    v8f acc0 = {}, acc1 = {};

    load_tile64(At[0], A, lda, mbase, 0, tid);
    load_tile64(Bt[0], W, ldw, nbase, 0, tid);
    for (int s = 0; s < S; ++s) {
        __syncthreads();
        if (s + 1 < S) {
            load_tile64(At[(s + 1) & 1], A, lda, mbase, (s + 1) * 32, tid);
            load_tile64(Bt[(s + 1) & 1], W, ldw, nbase, (s + 1) * 32, tid);
        }
        if (s + 2 < S) {
            prefetch_row(A, lda, mbase, (s + 2) * 32, tid);
            prefetch_row(W, ldw, nbase, (s + 2) * 32, tid);
        }
        const unsigned int* at = At[s & 1];
        const unsigned int* bt = Bt[s & 1];
        v16bf a  = build_a(at, m_off, lane);
        v16bf b0 = build_b(bt, n_off, lane);
        v16bf b1 = build_b(bt, n_off + 16, lane);
        acc0 = WMMA_BF16(a, b0, acc0);
        acc1 = WMMA_BF16(a, b1, acc1);
    }

    // C/D layout: VGPR r -> M = r (+8 for lanes 16-31), N = lane&15
    int col   = nbase + n_off + (lane & 15);
    int rbase = mbase + m_off + ((lane >> 4) * 8);
#pragma unroll
    for (int r = 0; r < 8; ++r) {
        size_t row = (size_t)(rbase + r);
        C[row * ldc + col]      = acc0[r];
        C[row * ldc + col + 16] = acc1[r];
    }
}

// ---------------------------------------------------------------------------
// Encoder: gates = x @ Wih^T (f-gate dead: h_prev=c_prev=0). Fuses i/g/o gate
// blocks (3 accumulators, shared A fragment) + sigmoid/tanh epilogue; writes
// hc = [hs ; cs] (4096 x 1024).  grid = (4096/64, 512/32)
// ---------------------------------------------------------------------------
__global__ void encoder_gates_wmma(const float* __restrict__ x,     // [4096,256]
                                   const float* __restrict__ Wih,   // [2048,256]
                                   const float* __restrict__ bias,  // [2048]
                                   float* __restrict__ hc) {        // [4096,1024]
    __shared__ unsigned int At[2][64 * TPITCH];
    __shared__ unsigned int Bt[2][3][32 * TPITCH];
    int tid  = threadIdx.x;
    int lane = tid & 31;
    int wv   = tid >> 5;
    int mbase = blockIdx.x * 64;
    int nb    = blockIdx.y * 32;         // within HE (0..511)
    int m_off = (wv & 3) * 16;
    int n_off = (wv >> 2) * 16;
    const int gate_row[3] = {0, 1024, 1536};   // i, g, o blocks of Wih

    v8f acc_i = {}, acc_g = {}, acc_o = {};

    load_tile64(At[0], x, 256, mbase, 0, tid);
#pragma unroll
    for (int g = 0; g < 3; ++g)
        load_tile32(Bt[0][g], Wih, 256, gate_row[g] + nb, 0, tid);

    for (int s = 0; s < 8; ++s) {
        __syncthreads();
        if (s + 1 < 8) {
            int k0 = (s + 1) * 32;
            load_tile64(At[(s + 1) & 1], x, 256, mbase, k0, tid);
#pragma unroll
            for (int g = 0; g < 3; ++g)
                load_tile32(Bt[(s + 1) & 1][g], Wih, 256, gate_row[g] + nb, k0, tid);
        }
        const unsigned int* at = At[s & 1];
        const unsigned int (*bts)[32 * TPITCH] = Bt[s & 1];
        v16bf a  = build_a(at, m_off, lane);
        v16bf bi = build_b(bts[0], n_off, lane);
        v16bf bg = build_b(bts[1], n_off, lane);
        v16bf bo = build_b(bts[2], n_off, lane);
        acc_i = WMMA_BF16(a, bi, acc_i);
        acc_g = WMMA_BF16(a, bg, acc_g);
        acc_o = WMMA_BF16(a, bo, acc_o);
    }

    int col   = nb + n_off + (lane & 15);        // 0..511 within HE
    int rbase = mbase + m_off + ((lane >> 4) * 8);
    float bi = bias[col];
    float bg = bias[1024 + col];
    float bo = bias[1536 + col];
#pragma unroll
    for (int r = 0; r < 8; ++r) {
        float gi = acc_i[r] + bi;
        float gg = acc_g[r] + bg;
        float go = acc_o[r] + bo;
        float cs = sigmoidf_(gi) * tanhf(gg);
        float hs = sigmoidf_(go) * tanhf(cs);
        size_t row = (size_t)(rbase + r);
        hc[row * 1024 + col]       = hs;
        hc[row * 1024 + 512 + col] = cs;
    }
}

// ---------------------------------------------------------------------------
// Fused decoder step: G = z_s @ Wih^T + h_in @ Whh^T (all four gates), then
// LSTM cell applied in-register.  h ping-pongs (cross-block RAW on h), c is
// updated in place (one owner block per element).
// Block tile 64(M=T) x 64(N within HD); wave tile 16x32 per gate.
// grid = (256/64, 512/64) = (4, 8).  24 K-stages (8 from z, 16 from h).
// ---------------------------------------------------------------------------
__global__ void decoder_step_wmma(const float* __restrict__ z_s,   // [256,256]
                                  const float* __restrict__ Wih,   // [2048,256]
                                  const float* __restrict__ Whh,   // [2048,512]
                                  const float* __restrict__ bias,  // [2048]
                                  const float* __restrict__ hin,   // [256,512]
                                  float* __restrict__ c,           // [256,512]
                                  float* __restrict__ hout,        // [256,512]
                                  float* __restrict__ ylast,       // [16,512]
                                  int step) {
    __shared__ unsigned int lds[2][5][64 * TPITCH];   // [buf][A, Bi, Bf, Bg, Bo]
    int tid  = threadIdx.x;
    int lane = tid & 31;
    int wv   = tid >> 5;
    int mbase = blockIdx.x * 64;         // T rows
    int nb    = blockIdx.y * 64;         // within HD (0..511)
    int m_off = (wv & 3) * 16;
    int n_off = (wv >> 2) * 32;

    v8f acc[4][2] = {};                  // [gate i,f,g,o][n-half]
    const int S = 24;                    // 8 stages K=256 (z), 16 stages K=512 (h)

    auto loadStage = [&](int s, int buf) {
        const float* A; const float* W; int lda, k0;
        if (s < 8) { A = z_s; W = Wih; lda = 256; k0 = s * 32; }
        else       { A = hin; W = Whh; lda = 512; k0 = (s - 8) * 32; }
        load_tile64(lds[buf][0], A, lda, mbase, k0, tid);
#pragma unroll
        for (int g = 0; g < 4; ++g)
            load_tile64(lds[buf][1 + g], W, lda, g * 512 + nb, k0, tid);
    };

    loadStage(0, 0);
    for (int s = 0; s < S; ++s) {
        __syncthreads();
        if (s + 1 < S) loadStage(s + 1, (s + 1) & 1);
        const unsigned int (*buf)[64 * TPITCH] = lds[s & 1];
        v16bf a = build_a(buf[0], m_off, lane);
#pragma unroll
        for (int g = 0; g < 4; ++g) {
            v16bf b0 = build_b(buf[1 + g], n_off, lane);
            v16bf b1 = build_b(buf[1 + g], n_off + 16, lane);
            acc[g][0] = WMMA_BF16(a, b0, acc[g][0]);
            acc[g][1] = WMMA_BF16(a, b1, acc[g][1]);
        }
    }

    int rbase = mbase + m_off + ((lane >> 4) * 8);
#pragma unroll
    for (int nh = 0; nh < 2; ++nh) {
        int col = nb + n_off + nh * 16 + (lane & 15);   // 0..511 within HD
        float bi = bias[col];
        float bf = bias[512 + col];
        float bg = bias[1024 + col];
        float bo = bias[1536 + col];
#pragma unroll
        for (int r = 0; r < 8; ++r) {
            int row = rbase + r;
            size_t idx = (size_t)row * 512 + col;
            float gi = acc[0][nh][r] + bi;
            float gf = acc[1][nh][r] + bf;
            float gg = acc[2][nh][r] + bg;
            float go = acc[3][nh][r] + bo;
            float cn = sigmoidf_(gf) * c[idx] + sigmoidf_(gi) * tanhf(gg);
            float hn = sigmoidf_(go) * tanhf(cn);
            c[idx]    = cn;
            hout[idx] = hn;
            if (row == 255) ylast[step * 512 + col] = hn;
        }
    }
}

// ---------------------------------------------------------------------------
// Attention: alpha[b,t,k] = v . tanh(w1hc[b,t] + w2x[b,k]); softmax over k;
// z[b,t,:] = beta @ x[b].  One block per (t,b); thread = k, then thread = i.
// ---------------------------------------------------------------------------
__global__ void attention_softmax_z(const float* __restrict__ w1hc,  // [4096,128]
                                    const float* __restrict__ w2x,   // [4096,128]
                                    const float* __restrict__ att_v, // [128]
                                    const float* __restrict__ x,     // [4096,256]
                                    float* __restrict__ z) {         // [4096,256]
    __shared__ float q[128], vv[128], beta[256], red[256];
    int t = blockIdx.x, b = blockIdx.y, tid = threadIdx.x;
    if (tid < 128) {
        q[tid]  = w1hc[(size_t)(b * 256 + t) * 128 + tid];
        vv[tid] = att_v[tid];
    }
    __syncthreads();

    const float* wr = w2x + (size_t)(b * 256 + tid) * 128;
    float s = 0.0f;
#pragma unroll 4
    for (int v = 0; v < 128; ++v) s += vv[v] * tanhf(q[v] + wr[v]);

    red[tid] = s; __syncthreads();
    for (int off = 128; off > 0; off >>= 1) {
        if (tid < off) red[tid] = fmaxf(red[tid], red[tid + off]);
        __syncthreads();
    }
    float mx = red[0]; __syncthreads();
    float e = __expf(s - mx);
    red[tid] = e; __syncthreads();
    for (int off = 128; off > 0; off >>= 1) {
        if (tid < off) red[tid] += red[tid + off];
        __syncthreads();
    }
    float inv = 1.0f / red[0]; __syncthreads();
    beta[tid] = e * inv; __syncthreads();

    float acc = 0.0f;
    const float* xb = x + (size_t)b * 256 * 256 + tid;
    for (int k = 0; k < 256; ++k) acc += beta[k] * xb[(size_t)k * 256];
    z[((size_t)b * 256 + t) * 256 + tid] = acc;
}

__global__ void zero_f32(float* __restrict__ p, int n) {
    int i = blockIdx.x * 256 + threadIdx.x;
    if (i < n) p[i] = 0.0f;
}

__global__ void final_fc(const float* __restrict__ ylast,  // [16,512]
                         const float* __restrict__ fcW,    // [2,512]
                         const float* __restrict__ fcb,    // [2]
                         float* __restrict__ out) {        // [16,2]
    int id = threadIdx.x;
    if (id < 32) {
        int b = id >> 1, j = id & 1;
        float s = 0.0f;
        for (int k = 0; k < 512; ++k) s += ylast[b * 512 + k] * fcW[j * 512 + k];
        out[b * 2 + j] = s + fcb[j];
    }
}

// ---------------------------------------------------------------------------
extern "C" void kernel_launch(void* const* d_in, const int* in_sizes, int n_in,
                              void* d_out, int out_size, void* d_ws, size_t ws_size,
                              hipStream_t stream) {
    (void)in_sizes; (void)n_in; (void)out_size; (void)ws_size;
    const float* x       = (const float*)d_in[0];
    const float* enc_Wih = (const float*)d_in[1];
    const float* enc_b   = (const float*)d_in[2];
    const float* att_W1  = (const float*)d_in[3];
    const float* att_W2  = (const float*)d_in[4];
    const float* att_v   = (const float*)d_in[5];
    const float* dec_Wih = (const float*)d_in[6];
    const float* dec_Whh = (const float*)d_in[7];
    const float* dec_b   = (const float*)d_in[8];
    const float* fc_W    = (const float*)d_in[9];
    const float* fc_b    = (const float*)d_in[10];

    float* ws    = (float*)d_ws;
    float* hc    = ws;                       // 4096*1024  [hs;cs]
    float* w1hc  = hc    + 4096 * 1024;      // 4096*128
    float* w2x   = w1hc  + 4096 * 128;       // 4096*128
    float* z     = w2x   + 4096 * 128;       // 4096*256
    float* h0    = z     + 4096 * 256;       // 256*512
    float* c     = h0    + 256 * 512;        // 256*512 (contiguous with h0)
    float* h1    = c     + 256 * 512;        // 256*512
    float* ylast = h1    + 256 * 512;        // 16*512

    // 1) encoder: fused i/g/o gate GEMM + activations -> hc
    encoder_gates_wmma<<<dim3(64, 16), 256, 0, stream>>>(x, enc_Wih, enc_b, hc);

    // 2) attention projections: w1hc = hc @ W1^T, w2x = x @ W2^T
    gemm_wT_wmma<<<dim3(64, 2), 256, 0, stream>>>(hc, 1024, att_W1, 1024,
                                                  w1hc, 128, 1024);
    gemm_wT_wmma<<<dim3(64, 2), 256, 0, stream>>>(x, 256, att_W2, 256,
                                                  w2x, 128, 256);

    // 3) scores + softmax + context
    attention_softmax_z<<<dim3(256, 16), 256, 0, stream>>>(w1hc, w2x, att_v, x, z);

    // 4) decoder LSTM scan over B, fully fused step kernel; h ping-pong
    zero_f32<<<(2 * 256 * 512 + 255) / 256, 256, 0, stream>>>(h0, 2 * 256 * 512);
    for (int b = 0; b < 16; ++b) {
        const float* hin = (b & 1) ? h1 : h0;
        float*       ho  = (b & 1) ? h0 : h1;
        decoder_step_wmma<<<dim3(4, 8), 256, 0, stream>>>(
            z + (size_t)b * 256 * 256, dec_Wih, dec_Whh, dec_b,
            hin, c, ho, ylast, b);
    }

    // 5) final projection [16,2]
    final_fc<<<1, 64, 0, stream>>>(ylast, fc_W, fc_b, (float*)d_out);
}
